// GeMAttention_35356170781143
// MI455X (gfx1250) — matrix operand order repaired and checked by reading, not compiled, not run on hardware
//
#include <hip/hip_runtime.h>
#include <hip/hip_bf16.h>
#include <math.h>

typedef __attribute__((ext_vector_type(16))) __bf16 v16bf;
typedef __attribute__((ext_vector_type(8)))  __bf16 v8bf;
typedef __attribute__((ext_vector_type(4)))  __bf16 v4bf;
typedef __attribute__((ext_vector_type(8)))  float  v8f;

#define HID   1024
#define NHEAD 16
#define AH    64
#define SEQ   2048
#define BATCH 2

// ---------------------------------------------------------------- TDM support
// This toolchain (clang-23 / therock-10.0) uses the 6-arg form:
//   tensor_load_to_lds(u32x4 g0, i32x8 g1, i32x4 g2, i32x4 g3, i32x8 extra, i32 cpol)
#if defined(__gfx1250__) && __has_builtin(__builtin_amdgcn_tensor_load_to_lds)
#define HAVE_TDM 1
#else
#define HAVE_TDM 0
#endif

typedef __attribute__((ext_vector_type(4))) unsigned int u32x4;
typedef __attribute__((ext_vector_type(8))) int          i32x8;
typedef __attribute__((ext_vector_type(4))) int          i32x4;

__device__ __forceinline__ void tdm_wait0() {
#if defined(__gfx1250__)
#if __has_builtin(__builtin_amdgcn_s_wait_tensorcnt)
  __builtin_amdgcn_s_wait_tensorcnt(0);
#else
  asm volatile("s_wait_tensorcnt 0x0" ::: "memory");
#endif
#endif
}

#if HAVE_TDM
// One TDM DMA of a 2D bf16 tile (tile_d0 contiguous elems x tile_d1 rows,
// row stride stride_elems) from global into LDS at lds_byte_off, with the
// TDM pad feature inserting pad after each (1<<pad_int_code)*8 bytes so LDS
// rows land on a bank-conflict-free padded stride.
__device__ __forceinline__ void tdm_load_2d_bf16(
    const __bf16* gsrc, unsigned int lds_byte_off,
    unsigned int tile_d0, unsigned int tile_d1, unsigned long long stride_elems,
    unsigned int pad_int_code, unsigned int pad_amt_code)
{
  unsigned long long ga = (unsigned long long)(uintptr_t)gsrc;
  u32x4 g0;
  g0[0] = 1u;                                   // count=1 (valid user descriptor)
  g0[1] = lds_byte_off;                         // lds_addr
  g0[2] = (unsigned int)ga;                     // global_addr[31:0]
  g0[3] = (unsigned int)((ga >> 32) & 0x01FFFFFFu) | (2u << 30); // addr[56:32], type=2
  i32x8 g1;
  unsigned int w0 = (1u << 16)                  // data_size=1 -> 2 bytes
                  | (1u << 20)                  // pad_enable
                  | (pad_int_code << 22)        // pad_interval
                  | (pad_amt_code << 25);       // pad_amount
  g1[0] = (int)w0;
  g1[1] = (int)((tile_d0 & 0xFFFFu) << 16);                 // tensor_dim0[15:0] (== tile)
  g1[2] = (int)(((tile_d0 >> 16) & 0xFFFFu) |
                ((tile_d1 & 0xFFFFu) << 16));               // tensor_dim0 hi, tensor_dim1 lo
  g1[3] = (int)(((tile_d1 >> 16) & 0xFFFFu) |
                ((tile_d0 & 0xFFFFu) << 16));               // tensor_dim1 hi, tile_dim0
  g1[4] = (int)(tile_d1 & 0xFFFFu);                         // tile_dim1 (tile_dim2 = 0)
  g1[5] = (int)(unsigned int)(stride_elems & 0xFFFFFFFFull);   // dim0_stride[31:0]
  g1[6] = (int)(unsigned int)((stride_elems >> 32) & 0xFFFFull); // dim0_stride[47:32]
  g1[7] = 0;
  i32x4 z4; z4[0] = z4[1] = z4[2] = z4[3] = 0;
  i32x8 z8;
#pragma unroll
  for (int i = 0; i < 8; ++i) z8[i] = 0;
  __builtin_amdgcn_tensor_load_to_lds(g0, g1, z4, z4, z8, 0);
}
#endif

__device__ __forceinline__ unsigned int lds_off_u32(const void* p) {
  // generic pointer to LDS: low 32 bits are the wave-relative LDS byte offset
  return (unsigned int)(uintptr_t)p;
}

// ---------------------------------------------------------------- helpers

__device__ __forceinline__ v16bf pack16(v8bf lo, v8bf hi) {
  v16bf r;
#pragma unroll
  for (int e = 0; e < 8; ++e) { r[e] = lo[e]; r[8 + e] = hi[e]; }
  return r;
}

__device__ __forceinline__ v8f wmma_bf16(v16bf a, v16bf b, v8f c) {
  return __builtin_amdgcn_wmma_f32_16x16x32_bf16(
      false, a, false, b, (short)0, c, false, false);
}

__device__ __forceinline__ float hmax16(float x) {
  x = fmaxf(x, __shfl_xor(x, 1, 32));
  x = fmaxf(x, __shfl_xor(x, 2, 32));
  x = fmaxf(x, __shfl_xor(x, 4, 32));
  x = fmaxf(x, __shfl_xor(x, 8, 32));
  return x;
}
__device__ __forceinline__ float hsum16(float x) {
  x += __shfl_xor(x, 1, 32);
  x += __shfl_xor(x, 2, 32);
  x += __shfl_xor(x, 4, 32);
  x += __shfl_xor(x, 8, 32);
  return x;
}

// ---------------------------------------------------------------- f32 -> bf16

__global__ __launch_bounds__(256) void k_cvt_bf16(const float* __restrict__ in,
                                                  __bf16* __restrict__ out, int n4) {
  int i = blockIdx.x * blockDim.x + threadIdx.x;
  if (i >= n4) return;
  float4 v = ((const float4*)in)[i];
  v4bf o;
  o[0] = (__bf16)v.x; o[1] = (__bf16)v.y; o[2] = (__bf16)v.z; o[3] = (__bf16)v.w;
  ((v4bf*)out)[i] = o;
}

// ---------------------------------------------------------------- GEMM: C = X @ W^T
// X:(M,K), W:(N,K) bf16 row-major; contraction along the fast dim of both maps
// transpose-free onto WMMA A/B fragments. Block tile 128x64, K-step 32, 8 waves.
// Tiles staged via TDM (double-buffered, DMA overlaps WMMA) with descriptor
// padding -> 80-byte LDS rows (conflict-free for the 16-row strided b128 reads).

template <int OUT_MODE>  // 0: bf16 store, 1: f32 store + bias
__global__ __launch_bounds__(256) void k_gemm_xwt(
    const __bf16* __restrict__ X, const __bf16* __restrict__ W,
    const float* __restrict__ bias, void* __restrict__ Cout,
    int M, int N, int K)
{
  __shared__ __bf16 sA[2][128][40];   // 32 elems + 8 pad = 80B rows
  __shared__ __bf16 sB[2][64][40];

  const int tid  = threadIdx.x;
  const int lane = tid & 31;
  const int w    = tid >> 5;
  const int wm   = w >> 1;     // 0..3
  const int wn   = w & 1;      // 0..1
  const int l15  = lane & 15;
  const int hi   = lane >> 4;

  const int n0 = blockIdx.x * 64;
  const int m0 = blockIdx.y * 128;

  v8f acc[2][2];
#pragma unroll
  for (int i = 0; i < 2; ++i)
#pragma unroll
    for (int j = 0; j < 2; ++j)
#pragma unroll
      for (int r = 0; r < 8; ++r) acc[i][j][r] = 0.f;

#if HAVE_TDM
  // ---- software-pipelined TDM tile loads: wave0 drives the DMA engine
  if (tid < 32) {
    tdm_load_2d_bf16(X + (size_t)m0 * K, lds_off_u32(&sA[0][0][0]),
                     32u, 128u, (unsigned long long)K, 3u, 3u); // 64B rows, +16B pad
    tdm_load_2d_bf16(W + (size_t)n0 * K, lds_off_u32(&sB[0][0][0]),
                     32u, 64u, (unsigned long long)K, 3u, 3u);
  }
  int buf = 0;
  for (int kk = 0; kk < K; kk += 32) {
    if (tid < 32) tdm_wait0();
    __syncthreads();                 // tile(kk) resident; other buffer free
    if (tid < 32 && kk + 32 < K) {
      tdm_load_2d_bf16(X + (size_t)m0 * K + kk + 32, lds_off_u32(&sA[buf ^ 1][0][0]),
                       32u, 128u, (unsigned long long)K, 3u, 3u);
      tdm_load_2d_bf16(W + (size_t)n0 * K + kk + 32, lds_off_u32(&sB[buf ^ 1][0][0]),
                       32u, 64u, (unsigned long long)K, 3u, 3u);
    }
#else
  const int brow = tid >> 2;
  const int bc8  = (tid & 3) * 8;
  int buf = 0;
  for (int kk = 0; kk < K; kk += 32) {
#pragma unroll
    for (int p = 0; p < 2; ++p) {
      int idx = p * 256 + tid;
      int row = idx >> 2;
      int c8  = (idx & 3) * 8;
      *(float4*)(&sA[0][row][c8]) =
          *(const float4*)(X + (size_t)(m0 + row) * K + kk + c8);
    }
    *(float4*)(&sB[0][brow][bc8]) =
        *(const float4*)(W + (size_t)(n0 + brow) * K + kk + bc8);
    __syncthreads();
#endif

    v16bf af[2], bfr[2];
#pragma unroll
    for (int i = 0; i < 2; ++i) {
      const __bf16* ap = &sA[buf][wm * 32 + i * 16 + l15][hi * 8];
      af[i] = pack16(*(const v8bf*)ap, *(const v8bf*)(ap + 16));
    }
#pragma unroll
    for (int j = 0; j < 2; ++j) {
      const __bf16* bp = &sB[buf][wn * 32 + j * 16 + l15][hi * 8];
      bfr[j] = pack16(*(const v8bf*)bp, *(const v8bf*)(bp + 16));
    }
#pragma unroll
    for (int i = 0; i < 2; ++i)
#pragma unroll
      for (int j = 0; j < 2; ++j)
        acc[i][j] = wmma_bf16(af[i], bfr[j], acc[i][j]);

#if HAVE_TDM
    buf ^= 1;
#else
    __syncthreads();
#endif
  }

#pragma unroll
  for (int i = 0; i < 2; ++i) {
#pragma unroll
    for (int j = 0; j < 2; ++j) {
      int col  = n0 + wn * 32 + j * 16 + l15;
      int rowb = m0 + wm * 32 + i * 16 + hi * 8;
      if (OUT_MODE == 0) {
        __bf16* C = (__bf16*)Cout;
#pragma unroll
        for (int r = 0; r < 8; ++r)
          C[(size_t)(rowb + r) * N + col] = (__bf16)acc[i][j][r];
      } else {
        float* C = (float*)Cout;
        float bc = bias[col];
#pragma unroll
        for (int r = 0; r < 8; ++r)
          C[(size_t)(rowb + r) * N + col] = acc[i][j][r] + bc;
      }
    }
  }
}

// ---------------------------------------------------------------- z / z_max / VE^T

__global__ __launch_bounds__(256) void k_zmax_vet(
    const float* __restrict__ v, const float* __restrict__ shift,
    const float* __restrict__ p_param, float* __restrict__ zmax_out,
    __bf16* __restrict__ vet)
{
  const int bd = blockIdx.x;
  const int b  = bd / HID;
  const int d  = bd - b * HID;
  const float* vcol = v + (size_t)b * SEQ * HID + d;

  const float sh  = shift[d];
  const float pp  = p_param[d];
  const float psn = (pp >= 0.f) ? 1.f : -1.f;
  const float pc  = psn * fminf(fmaxf(fabsf(pp), 1e-4f), 1e4f);

  float m = -1e30f;
  for (int k = threadIdx.x; k < SEQ; k += 256) {
    float z = pc * __logf(fmaxf(fabsf(vcol[(size_t)k * HID] + sh), 1e-10f));
    m = fmaxf(m, z);
  }
  __shared__ float red[256];
  red[threadIdx.x] = m;
  __syncthreads();
  for (int s = 128; s > 0; s >>= 1) {
    if (threadIdx.x < s) red[threadIdx.x] = fmaxf(red[threadIdx.x], red[threadIdx.x + s]);
    __syncthreads();
  }
  const float zm = red[0];
  if (threadIdx.x == 0) zmax_out[bd] = zm;

  __bf16* outcol = vet + (size_t)bd * SEQ;
  for (int k = threadIdx.x; k < SEQ; k += 256) {
    float z = pc * __logf(fmaxf(fabsf(vcol[(size_t)k * HID] + sh), 1e-10f));
    outcol[k] = (__bf16)__expf(z - zm);
  }
}

// ---------------------------------------------------------------- flash GeM attention
// Grid (SEQ/64, NHEAD, BATCH), 128 threads = 4 waves, wave owns 16 q rows.
// K / VE^T 64x64 tiles staged once per block via double-buffered TDM (fixes the
// 4x duplicated global traffic), 144-byte padded LDS rows. S = Q K^T (WMMA),
// online softmax via shfl halves, P relayout through wave-private LDS, O += P VE.

__global__ __launch_bounds__(128) void k_flash(
    const __bf16* __restrict__ q, const __bf16* __restrict__ kmat,
    const __bf16* __restrict__ vet, const float* __restrict__ zmax,
    const float* __restrict__ shift, const float* __restrict__ p_param,
    __bf16* __restrict__ e_out)
{
  const int lane = threadIdx.x & 31;
  const int w    = threadIdx.x >> 5;
  const int l15  = lane & 15;
  const int hi   = lane >> 4;
  const int b    = blockIdx.z;
  const int h    = blockIdx.y;
  const int q0   = blockIdx.x * 64;

  __shared__ __bf16 sK[2][64][72];     // 128B rows + 16B pad
  __shared__ __bf16 sV[2][64][72];
  __shared__ __bf16 pLDS[4][16][72];   // wave-private P relayout buffer

  const __bf16* kbase = kmat + (size_t)(b * SEQ) * HID + h * AH;
  const __bf16* vbase = vet + (size_t)(b * HID + h * AH) * SEQ;

  v16bf qf[2];
  {
    const __bf16* qp = q + (size_t)(b * SEQ + q0 + w * 16 + l15) * HID + h * AH;
#pragma unroll
    for (int t = 0; t < 2; ++t) {
      const __bf16* p0 = qp + t * 32 + hi * 8;
      qf[t] = pack16(*(const v8bf*)p0, *(const v8bf*)(p0 + 16));
    }
  }

  v8f o[4];
#pragma unroll
  for (int j = 0; j < 4; ++j)
#pragma unroll
    for (int r = 0; r < 8; ++r) o[j][r] = 0.f;
  float rowm[8], rowl[8];
#pragma unroll
  for (int r = 0; r < 8; ++r) { rowm[r] = -1e30f; rowl[r] = 0.f; }

  const float scale = 0.03125f;  // 1/sqrt(1024)

#if HAVE_TDM
  if (threadIdx.x < 32) {
    tdm_load_2d_bf16(kbase, lds_off_u32(&sK[0][0][0]),
                     64u, 64u, (unsigned long long)HID, 4u, 3u);
    tdm_load_2d_bf16(vbase, lds_off_u32(&sV[0][0][0]),
                     64u, 64u, (unsigned long long)SEQ, 4u, 3u);
  }
#endif
  int buf = 0;
  for (int k0 = 0; k0 < SEQ; k0 += 64) {
#if HAVE_TDM
    if (threadIdx.x < 32) tdm_wait0();
    __syncthreads();                 // tiles(k0) resident; other buffer free
    if (threadIdx.x < 32 && k0 + 64 < SEQ) {
      tdm_load_2d_bf16(kbase + (size_t)(k0 + 64) * HID, lds_off_u32(&sK[buf ^ 1][0][0]),
                       64u, 64u, (unsigned long long)HID, 4u, 3u);
      tdm_load_2d_bf16(vbase + (k0 + 64), lds_off_u32(&sV[buf ^ 1][0][0]),
                       64u, 64u, (unsigned long long)SEQ, 4u, 3u);
    }
#else
    // cooperative sync copy: 64x64 bf16 = 512 chunks of 16B, 4 per thread
#pragma unroll
    for (int c = 0; c < 4; ++c) {
      int idx = c * 128 + threadIdx.x;
      int row = idx >> 3;
      int c8  = (idx & 7) * 8;
      *(float4*)(&sK[0][row][c8]) = *(const float4*)(kbase + (size_t)(k0 + row) * HID + c8);
      *(float4*)(&sV[0][row][c8]) = *(const float4*)(vbase + (size_t)row * SEQ + k0 + c8);
    }
    __syncthreads();
#endif

    // ---- S = Q @ K^T
    v8f s[4];
#pragma unroll
    for (int j = 0; j < 4; ++j) {
#pragma unroll
      for (int r = 0; r < 8; ++r) s[j][r] = 0.f;
#pragma unroll
      for (int t = 0; t < 2; ++t) {
        const __bf16* kp = &sK[buf][j * 16 + l15][t * 32 + hi * 8];
        v16bf bfrag = pack16(*(const v8bf*)kp, *(const v8bf*)(kp + 16));
        s[j] = wmma_bf16(qf[t], bfrag, s[j]);
      }
    }
    // ---- online softmax
#pragma unroll
    for (int r = 0; r < 8; ++r) {
#pragma unroll
      for (int j = 0; j < 4; ++j) s[j][r] *= scale;
      float mx = fmaxf(fmaxf(s[0][r], s[1][r]), fmaxf(s[2][r], s[3][r]));
      mx = hmax16(mx);
      float nm = fmaxf(rowm[r], mx);
      float corr = __expf(rowm[r] - nm);
      rowm[r] = nm;
      rowl[r] *= corr;
#pragma unroll
      for (int j = 0; j < 4; ++j) o[j][r] *= corr;
#pragma unroll
      for (int j = 0; j < 4; ++j) s[j][r] = __expf(s[j][r] - nm);
      float rs = s[0][r] + s[1][r] + s[2][r] + s[3][r];
      rowl[r] += hsum16(rs);
    }
    // ---- P relayout (wave-private; same-wave DS ops are in-order)
#pragma unroll
    for (int j = 0; j < 4; ++j)
#pragma unroll
      for (int r = 0; r < 8; ++r)
        pLDS[w][hi * 8 + r][j * 16 + l15] = (__bf16)s[j][r];
    v16bf pf[2];
#pragma unroll
    for (int t = 0; t < 2; ++t) {
      const __bf16* pp = &pLDS[w][l15][t * 32 + hi * 8];
      pf[t] = pack16(*(const v8bf*)pp, *(const v8bf*)(pp + 16));
    }
    // ---- O += P @ VE
#pragma unroll
    for (int j = 0; j < 4; ++j) {
#pragma unroll
      for (int t = 0; t < 2; ++t) {
        const __bf16* vp = &sV[buf][j * 16 + l15][t * 32 + hi * 8];
        v16bf bfrag = pack16(*(const v8bf*)vp, *(const v8bf*)(vp + 16));
        o[j] = wmma_bf16(pf[t], bfrag, o[j]);
      }
    }
#if HAVE_TDM
    buf ^= 1;
#else
    __syncthreads();
#endif
  }

  // ---- fused GeM epilogue: e = exp((z_max + log(mean)) / p) - shift
#pragma unroll
  for (int j = 0; j < 4; ++j) {
    const int d     = h * AH + j * 16 + l15;
    const float zm  = zmax[b * HID + d];
    const float pp  = p_param[d];
    const float psn = (pp >= 0.f) ? 1.f : -1.f;
    const float pc  = psn * fminf(fmaxf(fabsf(pp), 1e-4f), 1e4f);
    const float sh  = shift[d];
#pragma unroll
    for (int r = 0; r < 8; ++r) {
      float mean = o[j][r] / rowl[r];
      float e = __expf((zm + __logf(mean)) / pc) - sh;
      e_out[(size_t)(b * SEQ + q0 + w * 16 + hi * 8 + r) * HID + d] = (__bf16)e;
    }
  }
}

// ---------------------------------------------------------------- launcher

extern "C" void kernel_launch(void* const* d_in, const int* in_sizes, int n_in,
                              void* d_out, int out_size, void* d_ws, size_t ws_size,
                              hipStream_t stream) {
  (void)in_sizes; (void)n_in; (void)out_size; (void)ws_size;
  const float* query   = (const float*)d_in[0];
  const float* context = (const float*)d_in[1];
  const float* Wq      = (const float*)d_in[2];
  const float* Wk      = (const float*)d_in[3];
  const float* Wv      = (const float*)d_in[4];
  const float* bv      = (const float*)d_in[5];
  const float* Wo      = (const float*)d_in[6];
  const float* bo      = (const float*)d_in[7];
  const float* shift   = (const float*)d_in[8];
  const float* p_param = (const float*)d_in[9];

  const int M = BATCH * SEQ;
  const size_t actE = (size_t)M * HID;
  const size_t wE   = (size_t)HID * HID;

  char* ws = (char*)d_ws;
  size_t off = 0;
  auto alloc = [&](size_t bytes) -> char* {
    char* p = ws + off;
    off += (bytes + 255) & ~(size_t)255;
    return p;
  };
  __bf16* qry_bf = (__bf16*)alloc(actE * 2);
  __bf16* ctx_bf = (__bf16*)alloc(actE * 2);
  __bf16* Wq_bf  = (__bf16*)alloc(wE * 2);
  __bf16* Wk_bf  = (__bf16*)alloc(wE * 2);
  __bf16* Wv_bf  = (__bf16*)alloc(wE * 2);
  __bf16* Wo_bf  = (__bf16*)alloc(wE * 2);
  __bf16* q_bf   = (__bf16*)alloc(actE * 2);
  __bf16* k_bf   = (__bf16*)alloc(actE * 2);
  float*  v_f32  = (float*)alloc(actE * 4);
  float*  zmax   = (float*)alloc((size_t)BATCH * HID * 4);
  __bf16* vet    = (__bf16*)alloc(actE * 2);
  __bf16* e_bf   = (__bf16*)alloc(actE * 2);

  const int n4a = (int)(actE / 4), n4w = (int)(wE / 4);
  k_cvt_bf16<<<(n4a + 255) / 256, 256, 0, stream>>>(query,   qry_bf, n4a);
  k_cvt_bf16<<<(n4a + 255) / 256, 256, 0, stream>>>(context, ctx_bf, n4a);
  k_cvt_bf16<<<(n4w + 255) / 256, 256, 0, stream>>>(Wq, Wq_bf, n4w);
  k_cvt_bf16<<<(n4w + 255) / 256, 256, 0, stream>>>(Wk, Wk_bf, n4w);
  k_cvt_bf16<<<(n4w + 255) / 256, 256, 0, stream>>>(Wv, Wv_bf, n4w);
  k_cvt_bf16<<<(n4w + 255) / 256, 256, 0, stream>>>(Wo, Wo_bf, n4w);

  dim3 ggrid(HID / 64, M / 128);
  k_gemm_xwt<0><<<ggrid, 256, 0, stream>>>(qry_bf, Wq_bf, nullptr, q_bf, M, HID, HID);
  k_gemm_xwt<0><<<ggrid, 256, 0, stream>>>(ctx_bf, Wk_bf, nullptr, k_bf, M, HID, HID);
  k_gemm_xwt<1><<<ggrid, 256, 0, stream>>>(ctx_bf, Wv_bf, bv, v_f32, M, HID, HID);

  k_zmax_vet<<<BATCH * HID, 256, 0, stream>>>(v_f32, shift, p_param, zmax, vet);

  dim3 fgrid(SEQ / 64, NHEAD, BATCH);
  k_flash<<<fgrid, 128, 0, stream>>>(q_bf, k_bf, vet, zmax, shift, p_param, e_bf);

  k_gemm_xwt<1><<<ggrid, 256, 0, stream>>>(e_bf, Wo_bf, bo, (float*)d_out, M, HID, HID);
}
